// CodedVTRBlock_28140625723931
// MI455X (gfx1250) — matrix operand order repaired and checked by reading, not compiled, not run on hardware
//
#include <hip/hip_runtime.h>
#include <hip/hip_bf16.h>

// ---------------- problem constants ----------------
constexpr int NPTS_ = 100000;
constexpr int P_    = 96;      // planes
constexpr int DIM_  = 192;     // P*EXP
constexpr int K_    = 27;      // taps
constexpr int D_    = 3;       // dilations
constexpr int TILES = NPTS_ / 16;   // 6250 exact

// ---------------- vector types (CDNA5 WMMA fragments) ----------------
typedef __attribute__((ext_vector_type(16))) __bf16 v16bf;
typedef __attribute__((ext_vector_type(8)))  __bf16 v8bf;
typedef __attribute__((ext_vector_type(8)))  float  v8f;

static __device__ __forceinline__ v16bf load16(const __bf16* p0, const __bf16* p1) {
    v8bf lo = *(const v8bf*)p0;
    v8bf hi = *(const v8bf*)p1;
    return __builtin_shufflevector(lo, hi, 0,1,2,3,4,5,6,7,8,9,10,11,12,13,14,15);
}

// ---------------- workspace layout (bytes) ----------------
constexpr size_t al256(size_t x) { return (x + 255) & ~size_t(255); }
constexpr size_t SZ_XB = size_t(NPTS_ + 1) * P_ * 2;      // x bf16, padded row
constexpr size_t SZ_VB = size_t(NPTS_) * P_ * 2;          // v bf16
constexpr size_t SZ_V2 = size_t(NPTS_ + 1) * DIM_ * 2;    // v2 bf16, padded row
constexpr size_t SZ_AG = size_t(NPTS_) * DIM_ * 2;        // agg bf16
constexpr size_t SZ_QS = size_t(NPTS_) * D_ * 8 * 4;      // qs fp32
constexpr size_t SZ_CH = SZ_QS;                           // choice fp32
constexpr size_t SZ_W1 = size_t(K_) * P_ * P_ * 2;
constexpr size_t SZ_W2 = size_t(P_) * DIM_ * 2;
constexpr size_t SZ_QW = size_t(D_) * K_ * P_ * 16 * 2;   // M padded 8->16
constexpr size_t SZ_OW = size_t(DIM_) * P_ * 2;
constexpr size_t OFF_XB = 0;
constexpr size_t OFF_VB = al256(OFF_XB + SZ_XB);
constexpr size_t OFF_V2 = al256(OFF_VB + SZ_VB);
constexpr size_t OFF_AG = al256(OFF_V2 + SZ_V2);
constexpr size_t OFF_QS = al256(OFF_AG + SZ_AG);
constexpr size_t OFF_CH = al256(OFF_QS + SZ_QS);
constexpr size_t OFF_W1 = al256(OFF_CH + SZ_CH);
constexpr size_t OFF_W2 = al256(OFF_W1 + SZ_W1);
constexpr size_t OFF_QW = al256(OFF_W2 + SZ_W2);
constexpr size_t OFF_OW = al256(OFF_QW + SZ_QW);

// prep element counts
constexpr long XB_N = long(NPTS_ + 1) * P_;
constexpr long W1_N = long(K_) * P_ * P_;
constexpr long W2_N = long(P_) * DIM_;
constexpr long QW_N = long(D_) * K_ * P_ * 16;
constexpr long OW_N = long(DIM_) * P_;
constexpr long PREP_N = XB_N + W1_N + W2_N + QW_N + OW_N + DIM_;

constexpr int W1_VEC = (P_ * P_) / 8;   // 1152 uint4 per tap tile

// mask bitmasks for sparse-pattern popcounts (MASKS in reference)
__constant__ unsigned c_mb[3][4] = {
    {0x0020CBu, 0x02C206u, 0x0005E1u, 0x0DA0000u},
    {0x701C00u, 0x30047Eu, 0x000FF8u, 0x1DE0000u},
    {0x482221u, 0x11098Au, 0x3041850u, 0x0884460u},
};

// ---------------- 0: convert / repack to bf16 ----------------
__global__ __launch_bounds__(256)
void k_prep(const float* __restrict__ x, const float* __restrict__ w1,
            const float* __restrict__ w2, const float* __restrict__ qw,
            const float* __restrict__ ow,
            __bf16* xb, __bf16* w1b, __bf16* w2b, __bf16* qwb, __bf16* owb, __bf16* v2b)
{
    long i = (long)blockIdx.x * 256 + threadIdx.x;
    if (i < XB_N) { xb[i] = (i < (long)NPTS_ * P_) ? (__bf16)x[i] : (__bf16)0.0f; return; }
    i -= XB_N;
    if (i < W1_N) { w1b[i] = (__bf16)w1[i]; return; }
    i -= W1_N;
    if (i < W2_N) { w2b[i] = (__bf16)w2[i]; return; }
    i -= W2_N;
    if (i < QW_N) {
        long col = i & 15, base = i >> 4;
        qwb[i] = (col < 8) ? (__bf16)qw[base * 8 + col] : (__bf16)0.0f;
        return;
    }
    i -= QW_N;
    if (i < OW_N) { owb[i] = (__bf16)ow[i]; return; }
    i -= OW_N;
    if (i < DIM_) { v2b[(size_t)NPTS_ * DIM_ + i] = (__bf16)0.0f; }
}

// ---------------- 1: value conv  v = relu(bn1(sum_k gather(x) @ W1[k])) ----------------
// Double-buffered LDS weight staging + prefetched neighbor indices +
// register-preloaded A/B fragments so WMMAs issue back-to-back.
__global__ __launch_bounds__(256)
void k_value_conv(const __bf16* __restrict__ xb, const __bf16* __restrict__ w1b,
                  const int* __restrict__ neis1,
                  const float* __restrict__ g, const float* __restrict__ b,
                  __bf16* __restrict__ vb)
{
    __shared__ __align__(16) __bf16 w1s[2][P_ * P_];   // 2 x 18 KB ping-pong
    const int tid  = threadIdx.x;
    const int wave = tid >> 5, lane = tid & 31;
    const int tile = blockIdx.x * 8 + wave;
    const bool active = tile < TILES;
    const int row = lane & 15;
    const int h8  = (lane >> 4) * 8;

    // prefetch all 27 neighbor indices (kills per-tap index->addr->load chain)
    int idxs[K_];
    #pragma unroll
    for (int k = 0; k < K_; ++k)
        idxs[k] = active ? neis1[(size_t)k * NPTS_ + tile * 16 + row] : NPTS_;

    // stage tap 0 weights
    {
        const uint4* src = (const uint4*)w1b;
        for (int i = tid; i < W1_VEC; i += 256) ((uint4*)w1s[0])[i] = src[i];
    }
    __syncthreads();

    v8f acc[6] = {};
    for (int k = 0; k < K_; ++k) {
        // ---- load next tap's weights into registers (overlaps WMMAs below) ----
        uint4 stg[5];
        const bool more = (k + 1 < K_);
        if (more) {
            const uint4* src = (const uint4*)(w1b + (size_t)(k + 1) * P_ * P_);
            #pragma unroll
            for (int j = 0; j < 5; ++j) {
                const int i = tid + j * 256;
                if (i < W1_VEC) stg[j] = src[i];
            }
        }
        // ---- compute current tap ----
        if (active) {
            const __bf16* xrow = xb + (size_t)idxs[k] * P_;
            const __bf16* bbase = w1s[k & 1];
            v16bf a[3];
            #pragma unroll
            for (int kc = 0; kc < 3; ++kc) {
                const __bf16* pa = xrow + kc * 32 + h8;
                a[kc] = load16(pa, pa + 16);
            }
            #pragma unroll
            for (int kc = 0; kc < 3; ++kc) {
                v16bf bfr[6];
                #pragma unroll
                for (int nt = 0; nt < 6; ++nt) {
                    const __bf16* pb = bbase + (kc * 32 + h8) * P_ + nt * 16 + row;
                    #pragma unroll
                    for (int s = 0; s < 16; ++s) bfr[nt][s] = pb[((s & 7) + (s >> 3) * 16) * P_];
                }
                #pragma unroll
                for (int nt = 0; nt < 6; ++nt)
                    acc[nt] = __builtin_amdgcn_wmma_f32_16x16x32_bf16(
                        false, a[kc], false, bfr[nt], (short)0, acc[nt], false, false);
            }
        }
        // ---- commit next tap's weights to the other LDS buffer ----
        if (more) {
            uint4* dst = (uint4*)w1s[(k + 1) & 1];
            #pragma unroll
            for (int j = 0; j < 5; ++j) {
                const int i = tid + j * 256;
                if (i < W1_VEC) dst[i] = stg[j];
            }
        }
        __syncthreads();
    }
    if (active) {
        #pragma unroll
        for (int nt = 0; nt < 6; ++nt) {
            const int col = nt * 16 + (lane & 15);
            const float gc = g[col], bc = b[col];
            #pragma unroll
            for (int r = 0; r < 8; ++r) {
                const int m = tile * 16 + r + h8;
                float v = acc[nt][r] * gc + bc;
                v = v > 0.f ? v : 0.f;
                vb[(size_t)m * P_ + col] = (__bf16)v;
            }
        }
    }
}

// ---------------- 2: expand  v2 = relu(bn2(v @ W2)) ----------------
__global__ __launch_bounds__(256)
void k_expand(const __bf16* __restrict__ vb, const __bf16* __restrict__ w2b,
              const float* __restrict__ g, const float* __restrict__ b,
              __bf16* __restrict__ v2b)
{
    __shared__ __align__(16) __bf16 w2s[P_ * DIM_];   // 36 KB staged once
    const int tid  = threadIdx.x;
    const int wave = tid >> 5, lane = tid & 31;
    const int tile = blockIdx.x * 8 + wave;
    const int row = lane & 15;
    const int h8  = (lane >> 4) * 8;
    for (int i = tid; i < (P_ * DIM_) / 8; i += 256)
        ((uint4*)w2s)[i] = ((const uint4*)w2b)[i];
    __syncthreads();
    if (tile >= TILES) return;

    const __bf16* arow = vb + (size_t)(tile * 16 + row) * P_;
    v16bf a[3];
    #pragma unroll
    for (int kc = 0; kc < 3; ++kc) {
        const __bf16* pa = arow + kc * 32 + h8;
        a[kc] = load16(pa, pa + 16);
    }
    v8f acc[12] = {};
    #pragma unroll
    for (int kc = 0; kc < 3; ++kc) {
        #pragma unroll
        for (int grp = 0; grp < 3; ++grp) {         // 4 B-fragments per group
            v16bf bfr[4];
            #pragma unroll
            for (int q = 0; q < 4; ++q) {
                const int nt = grp * 4 + q;
                const __bf16* pb = &w2s[(kc * 32 + h8) * DIM_ + nt * 16 + row];
                #pragma unroll
                for (int s = 0; s < 16; ++s) bfr[q][s] = pb[((s & 7) + (s >> 3) * 16) * DIM_];
            }
            #pragma unroll
            for (int q = 0; q < 4; ++q) {
                const int nt = grp * 4 + q;
                acc[nt] = __builtin_amdgcn_wmma_f32_16x16x32_bf16(
                    false, a[kc], false, bfr[q], (short)0, acc[nt], false, false);
            }
        }
    }
    #pragma unroll
    for (int nt = 0; nt < 12; ++nt) {
        const int col = nt * 16 + (lane & 15);
        const float gc = g[col], bc = b[col];
        #pragma unroll
        for (int r = 0; r < 8; ++r) {
            const int m = tile * 16 + r + h8;
            float v = acc[nt][r] * gc + bc;
            v = v > 0.f ? v : 0.f;
            v2b[(size_t)m * DIM_ + col] = (__bf16)v;
        }
    }
}

// ---------------- 3: query convs  qs[n,d,m] (M=8 padded to 16 cols) ----------------
__global__ __launch_bounds__(256)
void k_query(const __bf16* __restrict__ xb, const __bf16* __restrict__ qwb,
             const int* __restrict__ neis, float* __restrict__ qs)
{
    const int tid  = threadIdx.x;
    const int wave = tid >> 5, lane = tid & 31;
    const int tile = blockIdx.x * 8 + wave;
    if (tile >= TILES) return;
    const int row = lane & 15;
    const int h8  = (lane >> 4) * 8;

    v8f acc[3] = {};
    for (int d = 0; d < D_; ++d) {
        int idxs[K_];
        #pragma unroll
        for (int k = 0; k < K_; ++k)
            idxs[k] = neis[(size_t)(d * K_ + k) * NPTS_ + tile * 16 + row];
        for (int k = 0; k < K_; ++k) {
            const __bf16* xrow = xb + (size_t)idxs[k] * P_;
            v16bf a[3], bq[3];
            #pragma unroll
            for (int kc = 0; kc < 3; ++kc) {
                const __bf16* pa = xrow + kc * 32 + h8;
                a[kc] = load16(pa, pa + 16);
                const __bf16* pb = qwb + ((size_t)(d * K_ + k) * P_ + kc * 32 + h8) * 16 + row;
                #pragma unroll
                for (int s = 0; s < 16; ++s) bq[kc][s] = pb[((s & 7) + (s >> 3) * 16) * 16];
            }
            #pragma unroll
            for (int kc = 0; kc < 3; ++kc)
                acc[d] = __builtin_amdgcn_wmma_f32_16x16x32_bf16(
                    false, a[kc], false, bq[kc], (short)0, acc[d], false, false);
        }
    }
    const int m = lane & 15;
    if (m < 8) {
        for (int d = 0; d < D_; ++d)
            #pragma unroll
            for (int r = 0; r < 8; ++r)
                qs[((size_t)(tile * 16 + r + h8) * D_ + d) * 8 + m] = acc[d][r];
    }
}

// ---------------- 4: choice = softmax(qs/T) * double-softmax(sparse pattern) ----------------
__global__ __launch_bounds__(256)
void k_choice(const int* __restrict__ neis, const float* __restrict__ qs,
              float* __restrict__ choice)
{
    const int n = blockIdx.x * 256 + threadIdx.x;
    if (n >= NPTS_) return;
    unsigned ex[3];
    for (int d = 0; d < 3; ++d) {
        unsigned e = 0;
        for (int k = 0; k < K_; ++k)
            if (neis[(size_t)(d * K_ + k) * NPTS_ + n] < NPTS_) e |= (1u << k);
        ex[d] = e;
    }
    float sp[3][8];
    for (int d = 0; d < 3; ++d) {
        for (int m = 0; m < 8; ++m) sp[d][m] = 0.f;
        for (int mi = 0; mi < 4; ++mi) {
            unsigned keep = (~c_mb[d][mi]) & 0x7FFFFFFu;
            sp[d][mi] = (float)__popc(ex[d] & keep) / (float)__popc(keep);
        }
    }
    float p1[3][8];
    for (int d = 0; d < 3; ++d) {
        float t[8], mx = -1e30f, s = 0.f;
        for (int m = 0; m < 8; ++m) { t[m] = (sp[d][m] + 1e-3f) * 5.0f; mx = fmaxf(mx, t[m]); }
        for (int m = 0; m < 8; ++m) { t[m] = __expf(t[m] - mx); s += t[m]; }
        for (int m = 0; m < 8; ++m) p1[d][m] = t[m] / s;
    }
    float p2[3][8];
    for (int m = 0; m < 8; ++m) {
        float t[3], mx = -1e30f, s = 0.f;
        for (int d = 0; d < 3; ++d) { t[d] = (p1[d][m] + 1e-3f) * 5.0f; mx = fmaxf(mx, t[d]); }
        for (int d = 0; d < 3; ++d) { t[d] = __expf(t[d] - mx); s += t[d]; }
        for (int d = 0; d < 3; ++d) p2[d][m] = t[d] / s;
    }
    for (int d = 0; d < 3; ++d) {
        float t[8], mx = -1e30f, s = 0.f;
        for (int m = 0; m < 8; ++m) {
            t[m] = qs[((size_t)n * 3 + d) * 8 + m] * 0.70710678f;  // 1/sqrt(2)
            mx = fmaxf(mx, t[m]);
        }
        for (int m = 0; m < 8; ++m) { t[m] = __expf(t[m] - mx); s += t[m]; }
        for (int m = 0; m < 8; ++m)
            choice[((size_t)n * 3 + d) * 8 + m] = (t[m] / s) * p2[d][m];
    }
}

// ---------------- 5: agg = sum_d chconv(v2, neis[d], cb[d]) * expand(choice[:,:,d]) ----------------
__global__ __launch_bounds__(192)
void k_agg(const __bf16* __restrict__ v2b, const int* __restrict__ neis,
           const float* __restrict__ cb, const float* __restrict__ choice,
           __bf16* __restrict__ aggb)
{
    const int c = threadIdx.x;                 // channel 0..191 (coalesced gathers)
    const int n0 = blockIdx.x * 8;
    const int mdiv = c / 24;                   // DIM/M = 24
    for (int i = 0; i < 8; ++i) {
        const int n = n0 + i;
        if (n >= NPTS_) break;
        float a = 0.f;
        for (int d = 0; d < 3; ++d) {
            float s = 0.f;
            const float* cbd = cb + (size_t)d * K_ * DIM_ + c;
            const int* nd = neis + (size_t)d * K_ * NPTS_ + n;
            for (int k = 0; k < K_; ++k) {
                const int idx = nd[(size_t)k * NPTS_];
                s += cbd[(size_t)k * DIM_] * (float)v2b[(size_t)idx * DIM_ + c];
            }
            a += s * choice[((size_t)n * 3 + d) * 8 + mdiv];
        }
        aggb[(size_t)n * DIM_ + c] = (__bf16)a;
    }
}

// ---------------- 6: out = relu(bn_out(agg @ out_w)) + x ----------------
__global__ __launch_bounds__(256)
void k_out(const __bf16* __restrict__ aggb, const __bf16* __restrict__ owb,
           const float* __restrict__ g, const float* __restrict__ b,
           const float* __restrict__ x, float* __restrict__ out)
{
    __shared__ __align__(16) __bf16 ows[DIM_ * P_];   // 36 KB staged once
    const int tid  = threadIdx.x;
    const int wave = tid >> 5, lane = tid & 31;
    const int tile = blockIdx.x * 8 + wave;
    const int row = lane & 15;
    const int h8  = (lane >> 4) * 8;
    for (int i = tid; i < (DIM_ * P_) / 8; i += 256)
        ((uint4*)ows)[i] = ((const uint4*)owb)[i];
    __syncthreads();
    if (tile >= TILES) return;

    const __bf16* arow = aggb + (size_t)(tile * 16 + row) * DIM_;
    v8f acc[6] = {};
    #pragma unroll
    for (int kc = 0; kc < 6; ++kc) {
        const __bf16* pa = arow + kc * 32 + h8;
        v16bf a = load16(pa, pa + 16);
        v16bf bfr[6];
        #pragma unroll
        for (int nt = 0; nt < 6; ++nt) {
            const __bf16* pb = &ows[(kc * 32 + h8) * P_ + nt * 16 + row];
            #pragma unroll
            for (int s = 0; s < 16; ++s) bfr[nt][s] = pb[((s & 7) + (s >> 3) * 16) * P_];
        }
        #pragma unroll
        for (int nt = 0; nt < 6; ++nt)
            acc[nt] = __builtin_amdgcn_wmma_f32_16x16x32_bf16(
                false, a, false, bfr[nt], (short)0, acc[nt], false, false);
    }
    #pragma unroll
    for (int nt = 0; nt < 6; ++nt) {
        const int col = nt * 16 + (lane & 15);
        const float gc = g[col], bc = b[col];
        #pragma unroll
        for (int r = 0; r < 8; ++r) {
            const int m = tile * 16 + r + h8;
            float v = acc[nt][r] * gc + bc;
            v = v > 0.f ? v : 0.f;
            out[(size_t)m * P_ + col] = v + x[(size_t)m * P_ + col];
        }
    }
}

// ---------------- launcher ----------------
extern "C" void kernel_launch(void* const* d_in, const int* in_sizes, int n_in,
                              void* d_out, int out_size, void* d_ws, size_t ws_size,
                              hipStream_t stream)
{
    const float* x    = (const float*)d_in[0];
    const int*   neis = (const int*)d_in[1];
    const float* w1   = (const float*)d_in[2];
    const float* bn1g = (const float*)d_in[3];
    const float* bn1b = (const float*)d_in[4];
    const float* w2   = (const float*)d_in[5];
    const float* bn2g = (const float*)d_in[6];
    const float* bn2b = (const float*)d_in[7];
    const float* qw   = (const float*)d_in[8];
    const float* cb   = (const float*)d_in[9];
    const float* ow   = (const float*)d_in[10];
    const float* obng = (const float*)d_in[11];
    const float* obnb = (const float*)d_in[12];
    float* out = (float*)d_out;

    char* ws = (char*)d_ws;
    __bf16* xb   = (__bf16*)(ws + OFF_XB);
    __bf16* vb   = (__bf16*)(ws + OFF_VB);
    __bf16* v2b  = (__bf16*)(ws + OFF_V2);
    __bf16* aggb = (__bf16*)(ws + OFF_AG);
    float*  qsb  = (float*)(ws + OFF_QS);
    float*  chb  = (float*)(ws + OFF_CH);
    __bf16* w1b  = (__bf16*)(ws + OFF_W1);
    __bf16* w2b  = (__bf16*)(ws + OFF_W2);
    __bf16* qwb  = (__bf16*)(ws + OFF_QW);
    __bf16* owb  = (__bf16*)(ws + OFF_OW);

    const int* neis1 = neis + (size_t)1 * K_ * NPTS_;   // dilation index 1 for value conv

    k_prep<<<(int)((PREP_N + 255) / 256), 256, 0, stream>>>(x, w1, w2, qw, ow,
                                                            xb, w1b, w2b, qwb, owb, v2b);
    k_value_conv<<<(TILES + 7) / 8, 256, 0, stream>>>(xb, w1b, neis1, bn1g, bn1b, vb);
    k_expand<<<(TILES + 7) / 8, 256, 0, stream>>>(vb, w2b, bn2g, bn2b, v2b);
    k_query<<<(TILES + 7) / 8, 256, 0, stream>>>(xb, qwb, neis, qsb);
    k_choice<<<(NPTS_ + 255) / 256, 256, 0, stream>>>(neis, qsb, chb);
    k_agg<<<(NPTS_ + 7) / 8, 192, 0, stream>>>(v2b, neis, cb, chb, aggb);
    k_out<<<(TILES + 7) / 8, 256, 0, stream>>>(aggb, owb, obng, obnb, x, out);
}
